// DotProductAttention_223338299939
// MI455X (gfx1250) — matrix operand order repaired and checked by reading, not compiled, not run on hardware
//
#include <hip/hip_runtime.h>

// Problem constants (match reference)
#define BB 8
#define LL 2048
#define DD 64
#define BLOCK_M 64              // Q rows per workgroup (16 per wave)
#define BLOCK_N 64              // K/V rows per main-loop tile
#define NWAVES 4
#define NJT (LL / BLOCK_N)      // 32 tiles
#define LDS_STRIDE 66           // 64 + 2 pad (keeps 4B alignment, avoids bank conflicts)
#define RAW_STRIDE 68           // fp32 raw-tile row stride: 68*4 = 272B, 16B aligned

#if __has_builtin(__builtin_amdgcn_global_load_async_to_lds_b128) && \
    __has_builtin(__builtin_amdgcn_s_wait_asynccnt)
#define HAVE_ASYNC_LDS 1
#else
#define HAVE_ASYNC_LDS 0
#endif

typedef __attribute__((ext_vector_type(16))) __bf16 v16bf;
typedef __attribute__((ext_vector_type(8)))  float  v8f;

union AFrag {
    v16bf v;
    unsigned int w[8];
};

__device__ __forceinline__ unsigned short f2bf(float f) {
    union { float f; unsigned int u; } c;
    c.f = f;
    unsigned int u = c.u;
    u += 0x7fffu + ((u >> 16) & 1u);   // round-to-nearest-even
    return (unsigned short)(u >> 16);
}

__device__ __forceinline__ unsigned int pack2bf(float a, float b) {
    return (unsigned int)f2bf(a) | ((unsigned int)f2bf(b) << 16);
}

// A-matrix (16x32 bf16) K-offset for VGPR v, lane-half h (per CDNA5 ISA 7.12.2)
__device__ __forceinline__ int a_k0(int v, int h) {
    return ((v < 4) ? (2 * v) : (16 + 2 * (v - 4))) + 8 * h;
}

#if HAVE_ASYNC_LDS
typedef __attribute__((address_space(1))) void GVoid;
typedef __attribute__((address_space(3))) void LVoid;
typedef int v4i_raw __attribute__((vector_size(16)));   // matches builtin's pointee
typedef __attribute__((address_space(1))) v4i_raw* gvec_ptr;
typedef __attribute__((address_space(3))) v4i_raw* lvec_ptr;

__device__ __forceinline__ void async_ld16(const float* g, float* l) {
    // global_load_async_to_lds_b128: 16B/lane, memory -> LDS, tracked by ASYNCcnt
    __builtin_amdgcn_global_load_async_to_lds_b128(
        (gvec_ptr)(GVoid*)g, (lvec_ptr)(LVoid*)l, 0, 0);
}
#endif

__global__ __launch_bounds__(128)
void flash_attn_bf16_wmma(const float* __restrict__ Q,
                          const float* __restrict__ K,
                          float* __restrict__ O) {
    // bf16 staging: K-tile row-major (B-operand of Q*K^T), K-tile transposed
    // (V, B-operand of P*V), per-wave P scratch for C/D->A layout conversion.
    __shared__ unsigned short sKT[DD][LDS_STRIDE];        // sKT[d][krow]
    __shared__ unsigned short sKV[BLOCK_N][LDS_STRIDE];   // sKV[krow][d]
    __shared__ unsigned short sP[NWAVES][16][LDS_STRIDE]; // sP[w][m][kcol]
#if HAVE_ASYNC_LDS
    __shared__ float sRaw[2][BLOCK_N][RAW_STRIDE];        // double-buffered raw fp32 tile
#endif

    const int tid  = threadIdx.x;
    const int wave = tid >> 5;
    const int lane = tid & 31;
    const int half = lane >> 4;
    const int l16  = lane & 15;

    const int batch = blockIdx.x >> 5;        // 32 M-tiles per batch
    const int mtile = blockIdx.x & 31;
    const int qrow0 = mtile * BLOCK_M + wave * 16;

    const float* Qb = Q + (size_t)batch * LL * DD;
    const float* Kb = K + (size_t)batch * LL * DD;
    float*       Ob = O + (size_t)batch * LL * DD;

    // Cooperative staging assignment: 2 threads per K-row, 32 cols each
    const int lr = tid >> 1;
    const int lc = (tid & 1) * 32;

    // ---- Load Q A-fragments once, pre-scaled by 1/sqrt(D) ----
    AFrag aq[2];
    {
        const float qscale = 0.125f;  // 1/sqrt(64)
        const float* qrow = Qb + (size_t)(qrow0 + l16) * DD;
#pragma unroll
        for (int kk = 0; kk < 2; ++kk)
#pragma unroll
            for (int v = 0; v < 8; ++v) {
                int k0 = a_k0(v, half) + 32 * kk;
                float2 q2 = *(const float2*)(qrow + k0);
                aq[kk].w[v] = pack2bf(q2.x * qscale, q2.y * qscale);
            }
    }

    // ---- Online-softmax state ----
    v8f acc[4];
#pragma unroll
    for (int d = 0; d < 4; ++d) acc[d] = (v8f){};
    float m_i[8], l_i[8];
#pragma unroll
    for (int i = 0; i < 8; ++i) { m_i[i] = -3.0e38f; l_i[i] = 0.0f; }

#if HAVE_ASYNC_LDS
    // Prologue: stream tile 0 into LDS asynchronously (8 x b128 per thread)
#pragma unroll
    for (int cc = 0; cc < 32; cc += 4)
        async_ld16(Kb + (size_t)lr * DD + lc + cc, &sRaw[0][lr][lc + cc]);
#endif

    for (int j = 0; j < NJT; ++j) {
#if HAVE_ASYNC_LDS
        // Wait for this wave's async fills of tile j (each thread converts only
        // the chunks it fetched itself, so no barrier needed around sRaw).
        __builtin_amdgcn_s_wait_asynccnt(0);
#else
        const float* Kt = Kb + (size_t)j * BLOCK_N * DD;
#endif

        // ---- Convert tile j to bf16 in both layouts ----
#pragma unroll
        for (int cc = 0; cc < 32; cc += 4) {
            const int c = lc + cc;
#if HAVE_ASYNC_LDS
            float4 kv = *(const float4*)&sRaw[j & 1][lr][c];
#else
            float4 kv = *(const float4*)(Kt + (size_t)lr * DD + c);
#endif
            *(unsigned int*)&sKV[lr][c]     = pack2bf(kv.x, kv.y);
            *(unsigned int*)&sKV[lr][c + 2] = pack2bf(kv.z, kv.w);
            sKT[c    ][lr] = f2bf(kv.x);
            sKT[c + 1][lr] = f2bf(kv.y);
            sKT[c + 2][lr] = f2bf(kv.z);
            sKT[c + 3][lr] = f2bf(kv.w);
        }

        // ---- Kick off tile j+1 while we compute on tile j ----
        if (j + 1 < NJT) {
#if HAVE_ASYNC_LDS
            const float* Kn = Kb + (size_t)(j + 1) * BLOCK_N * DD;
#pragma unroll
            for (int cc = 0; cc < 32; cc += 4)
                async_ld16(Kn + (size_t)lr * DD + lc + cc,
                           &sRaw[(j + 1) & 1][lr][lc + cc]);
#else
            __builtin_prefetch(Kt + BLOCK_N * DD + (size_t)lr * DD + lc, 0, 0);
#endif
        }
        __syncthreads();

        // ---- S = (Q/sqrt(d)) * K^T : four 16x16 tiles, K chained over D=64 ----
        v8f s[4];
#pragma unroll
        for (int n = 0; n < 4; ++n) {
            v8f c8 = (v8f){};
#pragma unroll
            for (int kk = 0; kk < 2; ++kk) {
                AFrag bk;
#pragma unroll
                for (int v = 0; v < 8; ++v)
                    bk.w[v] = *(const unsigned int*)&sKV[n * 16 + l16][kk * 32 + a_k0(v, half)];
                c8 = __builtin_amdgcn_wmma_f32_16x16x32_bf16(
                         false, aq[kk].v, false, bk.v, (short)0, c8, false, false);
            }
            s[n] = c8;
        }

        // ---- Online softmax. Row(i) = i + 8*half lives in one 16-lane half. ----
        float mnew[8], rescale[8], rsum[8];
#pragma unroll
        for (int i = 0; i < 8; ++i) {
            float mx = fmaxf(fmaxf(s[0][i], s[1][i]), fmaxf(s[2][i], s[3][i]));
#pragma unroll
            for (int x = 8; x >= 1; x >>= 1)
                mx = fmaxf(mx, __shfl_xor(mx, x, 32));
            mnew[i]    = fmaxf(m_i[i], mx);
            rescale[i] = __expf(m_i[i] - mnew[i]);
            m_i[i]     = mnew[i];
            rsum[i]    = 0.0f;
        }
#pragma unroll
        for (int n = 0; n < 4; ++n)
#pragma unroll
            for (int i = 0; i < 8; ++i) {
                float p = __expf(s[n][i] - mnew[i]);
                rsum[i] += p;
                sP[wave][i + 8 * half][n * 16 + l16] = f2bf(p);
            }
#pragma unroll
        for (int i = 0; i < 8; ++i) {
#pragma unroll
            for (int x = 8; x >= 1; x >>= 1)
                rsum[i] += __shfl_xor(rsum[i], x, 32);
            l_i[i] = l_i[i] * rescale[i] + rsum[i];
        }
#pragma unroll
        for (int d = 0; d < 4; ++d)
#pragma unroll
            for (int i = 0; i < 8; ++i)
                acc[d][i] = acc[d][i] * rescale[i];

        // ---- O += P * V (V = K tile), P re-read from LDS in A layout ----
        AFrag ap[2];
#pragma unroll
        for (int kk = 0; kk < 2; ++kk)
#pragma unroll
            for (int v = 0; v < 8; ++v) {
                int k0 = a_k0(v, half) + 32 * kk;
                ap[kk].w[v] = *(const unsigned int*)&sP[wave][l16][k0];
            }
#pragma unroll
        for (int d = 0; d < 4; ++d)
#pragma unroll
            for (int kk = 0; kk < 2; ++kk) {
                AFrag bv;
#pragma unroll
                for (int v = 0; v < 8; ++v)
                    bv.w[v] = *(const unsigned int*)&sKT[d * 16 + l16][kk * 32 + a_k0(v, half)];
                acc[d] = __builtin_amdgcn_wmma_f32_16x16x32_bf16(
                             false, ap[kk].v, false, bv.v, (short)0, acc[d], false, false);
            }
        __syncthreads();  // compute(j) done before convert(j+1) overwrites sKT/sKV
    }

    // ---- Epilogue: normalize and store ----
#pragma unroll
    for (int i = 0; i < 8; ++i) {
        const float inv = 1.0f / l_i[i];
        const int row = qrow0 + i + 8 * half;
#pragma unroll
        for (int d = 0; d < 4; ++d)
            Ob[(size_t)row * DD + d * 16 + l16] = acc[d][i] * inv;
    }
}

extern "C" void kernel_launch(void* const* d_in, const int* in_sizes, int n_in,
                              void* d_out, int out_size, void* d_ws, size_t ws_size,
                              hipStream_t stream) {
    (void)in_sizes; (void)n_in; (void)d_ws; (void)ws_size; (void)out_size;
    const float* q = (const float*)d_in[0];
    const float* k = (const float*)d_in[1];
    float* out = (float*)d_out;

    dim3 grid(BB * (LL / BLOCK_M));  // 8 * 32 = 256 workgroups
    dim3 block(NWAVES * 32);         // 4 waves of 32
    flash_attn_bf16_wmma<<<grid, block, 0, stream>>>(q, k, out);
}
